// ScaledDotProduct_29343216566706
// MI455X (gfx1250) — matrix operand order repaired and checked by reading, not compiled, not run on hardware
//
#include <hip/hip_runtime.h>

typedef __attribute__((ext_vector_type(16))) __bf16 v16bf;
typedef __attribute__((ext_vector_type(8)))  float  v8f;

#define B_  2
#define H_  16
#define S_  2048
#define DK_ 128
#define DV_ 128

// LDS geometry (bf16 units)
#define K_ROW_BF   136              // 128 + 8 pad, 272B row (16B aligned: 17 * uint4)
#define K_ROW_U4   17
#define VT_ROW_BF  48               // 32 + 16 pad, 96B row (6 * uint4)
#define VT_ROW_U4  6

union F4 { float4 v; float f[4]; };

__global__ __launch_bounds__(256) void flash_attn_bf16_wmma(
    const float* __restrict__ Q, const float* __restrict__ K,
    const float* __restrict__ V, const int* __restrict__ Msk,
    float* __restrict__ O)
{
    const int tid  = threadIdx.x;
    const int lane = tid & 31;
    const int wave = tid >> 5;
    const int half = lane >> 4;      // 0 or 1
    const int ln   = lane & 15;

    const int bh   = blockIdx.x >> 4;          // 0..31 (b*H+h)
    const int qblk = blockIdx.x & 15;          // 16 q-blocks of 128 rows per head
    const int q_base = (qblk * 8 + wave) * 16; // this wave's 16 query rows

    const float* Qh = Q   + (size_t)bh * S_ * DK_;
    const float* Kh = K   + (size_t)bh * S_ * DK_;
    const float* Vh = V   + (size_t)bh * S_ * DV_;
    const int*   Mh = Msk + (size_t)bh * S_ * S_;
    float*       Oh = O   + (size_t)bh * S_ * DV_;

    __shared__ uint4 ldsK[32 * K_ROW_U4];    // K block, row-major bf16 [32][136]
    __shared__ uint4 ldsV[128 * VT_ROW_U4];  // V block, TRANSPOSED bf16 [128][48] = [d][key]

    // ---- Q fragments: B-operand of S^T = K * Q^T. B[k][n]=Q[q_base+n][k].
    // lane n=ln, element e -> feat = e + 16*half + 32*f  (contiguous 16 floats)
    v16bf qf[4];
    {
        const float* qrow = Qh + (size_t)(q_base + ln) * DK_;
        #pragma unroll
        for (int f = 0; f < 4; ++f) {
            const int f0 = half * 16 + f * 32;
            union { v16bf v; __bf16 e[16]; } cv;
            #pragma unroll
            for (int i = 0; i < 16; i += 4) {
                float4 t = *(const float4*)(qrow + f0 + i);
                cv.e[i+0] = (__bf16)t.x; cv.e[i+1] = (__bf16)t.y;
                cv.e[i+2] = (__bf16)t.z; cv.e[i+3] = (__bf16)t.w;
            }
            qf[f] = cv.v;
        }
    }

    v8f out[8];
    {
        v8f z = {};
        #pragma unroll
        for (int i = 0; i < 8; ++i) out[i] = z;
    }
    float m_i = -3.0e38f;   // running row max, base-2 domain (per query = ln, both halves)
    float l_i = 0.0f;       // running row sum
    // fold softmax scale and log2(e) into one constant applied to raw scores:
    // p = exp(s/sqrt(d) - m) = exp2(s * (log2e/sqrt(d)) - m2)
    const float c1 = 0.1275174512595276f;   // (1/sqrt(128)) * log2(e)

    // ---- software-pipelined K/V prefetch registers
    const int krow = tid >> 3;            // 0..31  (K row)
    const int kseg = (tid & 7) * 16;      // 0..112 (K feature segment)
    const int vq   = tid & 7;             // key quad: keys 4*vq..4*vq+3
    const int vd   = (tid >> 3) * 4;      // d segment: 4 d values
    F4 kpre[4], vpre[4];

    auto loadKV = [&](int kb) {
        const float* ks = Kh + (size_t)(kb + krow) * DK_ + kseg;
        #pragma unroll
        for (int i = 0; i < 4; ++i) kpre[i].v = *(const float4*)(ks + i * 4);
        #pragma unroll
        for (int j = 0; j < 4; ++j)
            vpre[j].v = *(const float4*)(Vh + (size_t)(kb + vq * 4 + j) * DV_ + vd);
    };

    loadKV(0);

    for (int kb = 0; kb < S_; kb += 32) {
        // ---- stage current block into LDS (convert f32 -> bf16)
        {
            union { __bf16 e[16]; uint4 u[2]; } ck;
            #pragma unroll
            for (int j = 0; j < 4; ++j) {
                ck.e[j*4+0] = (__bf16)kpre[j].f[0]; ck.e[j*4+1] = (__bf16)kpre[j].f[1];
                ck.e[j*4+2] = (__bf16)kpre[j].f[2]; ck.e[j*4+3] = (__bf16)kpre[j].f[3];
            }
            const int u4 = krow * K_ROW_U4 + (kseg >> 3);
            ldsK[u4]     = ck.u[0];
            ldsK[u4 + 1] = ck.u[1];

            // V: 4x4 in-register transpose, 4 x ds_store_b64 (keys contiguous in VshT row)
            __bf16* vt = (__bf16*)ldsV;
            #pragma unroll
            for (int i = 0; i < 4; ++i) {
                union { __bf16 e[4]; uint2 u; } cv;
                cv.e[0] = (__bf16)vpre[0].f[i]; cv.e[1] = (__bf16)vpre[1].f[i];
                cv.e[2] = (__bf16)vpre[2].f[i]; cv.e[3] = (__bf16)vpre[3].f[i];
                *(uint2*)(vt + (size_t)(vd + i) * VT_ROW_BF + vq * 4) = cv.u;
            }
        }
        __syncthreads();

        // ---- prefetch NEXT block's K/V (overlaps with WMMA + softmax below)
        if (kb + 32 < S_) loadKV(kb + 32);

        // ---- S^T tiles: acc_t element r: key = kb + 16*t + r + 8*half, query = q_base + ln
        v8f acc0 = {}, acc1 = {};
        #pragma unroll
        for (int f = 0; f < 4; ++f) {
            const int c1i = 2 * half + f * 4;       // uint4 offset of chunk1 in K row
            const int c2i = 2 * half + 1 + f * 4;   // uint4 offset of chunk2 in K row
            union { v16bf v; uint4 u[2]; } a0, a1;
            a0.u[0] = ldsK[ln * K_ROW_U4 + c1i];
            a0.u[1] = ldsK[ln * K_ROW_U4 + c2i];
            a1.u[0] = ldsK[(16 + ln) * K_ROW_U4 + c1i];
            a1.u[1] = ldsK[(16 + ln) * K_ROW_U4 + c2i];
            acc0 = __builtin_amdgcn_wmma_f32_16x16x32_bf16(false, a0.v, false, qf[f],
                                                           (short)0, acc0, false, false);
            acc1 = __builtin_amdgcn_wmma_f32_16x16x32_bf16(false, a1.v, false, qf[f],
                                                           (short)0, acc1, false, false);
        }

        // ---- mask: element r: tile0 key = kb + r + 8*half ; tile1 key +16
        const int* mrow = Mh + (size_t)(q_base + ln) * S_ + kb + half * 8;
        int4 m0a = *(const int4*)(mrow);
        int4 m0b = *(const int4*)(mrow + 4);
        int4 m1a = *(const int4*)(mrow + 16);
        int4 m1b = *(const int4*)(mrow + 20);
        int mk0[8] = { m0a.x, m0a.y, m0a.z, m0a.w, m0b.x, m0b.y, m0b.z, m0b.w };
        int mk1[8] = { m1a.x, m1a.y, m1a.z, m1a.w, m1b.x, m1b.y, m1b.z, m1b.w };

        // ---- online softmax in base-2 domain (per query = ln)
        float s0[8], s1[8];
        float tmax = -3.0e38f;
        #pragma unroll
        for (int r = 0; r < 8; ++r) {
            float a = acc0[r] * c1; if (mk0[r] == 1) a = -1.0e9f;
            float b = acc1[r] * c1; if (mk1[r] == 1) b = -1.0e9f;
            s0[r] = a; s1[r] = b;
            tmax = fmaxf(tmax, fmaxf(a, b));
        }
        tmax = fmaxf(tmax, __shfl_xor(tmax, 16, 32));   // combine key-halves
        const float m_new = fmaxf(m_i, tmax);
        const float corr  = __builtin_amdgcn_exp2f(m_i - m_new);
        float rsum = 0.0f;
        #pragma unroll
        for (int r = 0; r < 8; ++r) {
            s0[r] = __builtin_amdgcn_exp2f(s0[r] - m_new); rsum += s0[r];
            s1[r] = __builtin_amdgcn_exp2f(s1[r] - m_new); rsum += s1[r];
        }
        rsum += __shfl_xor(rsum, 16, 32);
        l_i = l_i * corr + rsum;
        m_i = m_new;

        // ---- rescale out accumulators: element r holds query (r + 8*half)
        float corr_e[8];
        #pragma unroll
        for (int r = 0; r < 8; ++r) corr_e[r] = __shfl(corr, r + 8 * half, 32);
        #pragma unroll
        for (int c = 0; c < 8; ++c)
            #pragma unroll
            for (int r = 0; r < 8; ++r) out[c][r] *= corr_e[r];

        // ---- pack P as bf16 A-fragment (S^T layout == A layout, no lane permutes)
        union { v16bf v; __bf16 e[16]; } pf;
        #pragma unroll
        for (int r = 0; r < 8; ++r) {
            pf.e[r]     = (__bf16)s0[r];   // k = r + 8*half
            pf.e[r + 8] = (__bf16)s1[r];   // k = 16 + r + 8*half
        }

        // ---- P(16x32) x V(32x128): 8 WMMAs over d-chunks
        #pragma unroll
        for (int c = 0; c < 8; ++c) {
            const int d = c * 16 + ln;     // B-frag lane = d column
            union { v16bf v; uint4 u[2]; } bf;
            bf.u[0] = ldsV[d * VT_ROW_U4 + half];          // keys 8*half + 0..7
            bf.u[1] = ldsV[d * VT_ROW_U4 + half + 2];      // keys 16 + 8*half + 0..7
            out[c] = __builtin_amdgcn_wmma_f32_16x16x32_bf16(false, pf.v, false, bf.v,
                                                             (short)0, out[c], false, false);
        }
        __syncthreads();   // LDS reads done before next iteration's stores
    }

    // ---- finalize: divide by row sum, store fp32
    const float linv = __builtin_amdgcn_rcpf(l_i);
    float linv_e[8];
    #pragma unroll
    for (int r = 0; r < 8; ++r) linv_e[r] = __shfl(linv, r + 8 * half, 32);

    #pragma unroll
    for (int c = 0; c < 8; ++c) {
        #pragma unroll
        for (int r = 0; r < 8; ++r) {
            const int q = q_base + r + 8 * half;
            Oh[(size_t)q * DV_ + c * 16 + ln] = out[c][r] * linv_e[r];
        }
    }
}

extern "C" void kernel_launch(void* const* d_in, const int* in_sizes, int n_in,
                              void* d_out, int out_size, void* d_ws, size_t ws_size,
                              hipStream_t stream) {
    const float* q = (const float*)d_in[0];
    const float* k = (const float*)d_in[1];
    const float* v = (const float*)d_in[2];
    const int* msk = (const int*)d_in[3];
    float* out = (float*)d_out;

    // 32 (b,h) heads * 16 query-blocks of 128 rows = 512 workgroups, 256 threads (8 waves)
    dim3 grid(B_ * H_ * 16);
    dim3 block(256);
    flash_attn_bf16_wmma<<<grid, block, 0, stream>>>(q, k, v, msk, out);
}
